// GraphAttentionLayer_953482740190
// MI455X (gfx1250) — matrix operand order repaired and checked by reading, not compiled, not run on hardware
//
#include <hip/hip_runtime.h>
#include <math.h>

// GAT layer fused kernel for MI455X (gfx1250, wave32).
//
// h = inp@W; e = lrelu(f1_i + f2_j) masked by adj; row softmax; out = elu(P@h).
// adj (256 MB int32, streamed once, non-temporal) dominates HBM traffic:
// floor ~11 us @ 23.3 TB/s. Ht (1 MB f16) + f2 (32 KB) stay L2-resident.
// lrelu is monotone => m_i = lrelu(f1_i + max_j f2_j) is a valid softmax
// bound computable up-front => single-pass accumulate, no online rescaling.
// K-split over 4 j-slices for ~8192 waves in flight (latency-BW product
// needs ~14 MB of outstanding loads).

typedef __attribute__((ext_vector_type(16))) _Float16 v16h;
typedef __attribute__((ext_vector_type(8)))  _Float16 v8h;
typedef __attribute__((ext_vector_type(8)))  float    v8f;
typedef __attribute__((ext_vector_type(4)))  int      v4i;
typedef __attribute__((ext_vector_type(4)))  float    v4f;

#define GAT_ALPHA 0.2f
#define GAT_N     8192
#define GAT_INF   128
#define GAT_OUTF  64
#define GAT_SPLIT 4
#define GAT_JSEG  (GAT_N / GAT_SPLIT)   // 2048, power of two

// ---------------------------------------------------------------------------
// Kernel 1: h = inp @ W (fp32), plus Ht = h^T in fp16 so the attention
// kernel's B fragments are two contiguous 16 B loads per lane.
// ---------------------------------------------------------------------------
__global__ void __launch_bounds__(256)
gat_h_kernel(const float* __restrict__ inp, const float* __restrict__ W,
             float* __restrict__ h, _Float16* __restrict__ ht) {
    int idx = blockIdx.x * 256 + threadIdx.x;    // N*OUTF threads
    int row = idx >> 6;
    int col = idx & 63;
    const float* ir = inp + (size_t)row * GAT_INF;
    float acc = 0.f;
#pragma unroll 8
    for (int k = 0; k < GAT_INF; ++k)
        acc = __builtin_fmaf(ir[k], W[k * GAT_OUTF + col], acc);
    h[(size_t)row * GAT_OUTF + col] = acc;
    ht[(size_t)col * GAT_N + row]   = (_Float16)acc;
}

// ---------------------------------------------------------------------------
// Kernel 2: f1 = h @ a[:64], f2 = h @ a[64:]
// ---------------------------------------------------------------------------
__global__ void __launch_bounds__(256)
gat_f_kernel(const float* __restrict__ h, const float* __restrict__ a,
             float* __restrict__ f1, float* __restrict__ f2) {
    int row = blockIdx.x * 256 + threadIdx.x;
    const float* hr = h + (size_t)row * GAT_OUTF;
    float s1 = 0.f, s2 = 0.f;
#pragma unroll 8
    for (int c = 0; c < GAT_OUTF; ++c) {
        float hv = hr[c];
        s1 = __builtin_fmaf(hv, a[c], s1);
        s2 = __builtin_fmaf(hv, a[GAT_OUTF + c], s2);
    }
    f1[row] = s1;
    f2[row] = s2;
}

// ---------------------------------------------------------------------------
// Kernel 3: gmax = max_j f2[j]
// ---------------------------------------------------------------------------
__global__ void __launch_bounds__(256)
gat_gmax_kernel(const float* __restrict__ f2, float* __restrict__ gmax) {
    __shared__ float red[256];
    float m = -3.4e38f;
    for (int i = threadIdx.x; i < GAT_N; i += 256) m = fmaxf(m, f2[i]);
    red[threadIdx.x] = m;
    __syncthreads();
    for (int s = 128; s > 0; s >>= 1) {
        if (threadIdx.x < s)
            red[threadIdx.x] = fmaxf(red[threadIdx.x], red[threadIdx.x + s]);
        __syncthreads();
    }
    if (threadIdx.x == 0) *gmax = red[0];
}

// ---------------------------------------------------------------------------
// Kernel 4: fused masked-softmax-matmul partial pass.
// grid = (N/16 row tiles, GAT_SPLIT j-slices); block = 128 threads (4 waves).
// Wave w owns output columns [16w,16w+16). Per K-step of 32: build P
// (16x32 f16, ISA A-fragment layout) from adj/f1/f2, load B (32x16 f16)
// from Ht, one v_wmma_f32_16x16x32_f16. Partials go to workspace.
// ---------------------------------------------------------------------------
__device__ __forceinline__ float gat_p(int av, float f2v, float f1r, float mi) {
    float e = f1r + f2v;
    e = (e > 0.f) ? e : GAT_ALPHA * e;
    float p = __expf(e - mi);
    return (av != 0) ? p : 0.f;
}

__global__ void __launch_bounds__(128)
gat_attn_kernel(const int* __restrict__ adj, const _Float16* __restrict__ ht,
                const float* __restrict__ f1, const float* __restrict__ f2,
                const float* __restrict__ gmaxp,
                float* __restrict__ accbuf, float* __restrict__ denbuf) {
    const int lane  = threadIdx.x & 31;
    const int wave  = threadIdx.x >> 5;
    const int laneL = lane & 15;
    const int hi    = lane >> 4;
    const int i0    = blockIdx.x * 16;           // row tile
    const int c0    = wave * 16;                 // this wave's output columns
    const int jbeg  = blockIdx.y * GAT_JSEG;     // this block's j slice

    // A-fragment (16-bit A 16x32, ISA 7.12.2):
    //   lanes 0-15 : M=lane,    K = {0..7, 16..23}
    //   lanes16-31 : M=lane-16, K = {8..15, 24..31}
    const int row = i0 + laneL;
    const int kA  = hi * 8;

    const float gmax = *gmaxp;
    const float f1r  = f1[row];
    float mt = f1r + gmax;
    const float mi = (mt > 0.f) ? mt : GAT_ALPHA * mt;   // row softmax bound

    const int*      arow = adj + (size_t)row * GAT_N + jbeg;
    // B-fragment: lane holds column c0+laneL, K = hi*8 + {0..7, 16..23}:
    // two 16 B runs in Ht.
    const _Float16* bcol = ht + (size_t)(c0 + laneL) * GAT_N + jbeg + hi * 8;

    v8f   c   = {};
    float den = 0.f;

    for (int j0 = 0; j0 < GAT_JSEG; j0 += 32) {
        // Unconditional prefetch, wrapped inside this block's slice
        // (prefetch has no side effects; avoids a per-iteration branch).
        __builtin_prefetch(arow + ((j0 + 2048) & (GAT_JSEG - 1)), 0, 3);

        // adj: one-shot 256 MB stream -> non-temporal so Ht/f2 stay in L2
        v4i q0 = __builtin_nontemporal_load((const v4i*)(arow + j0 + kA));
        v4i q1 = __builtin_nontemporal_load((const v4i*)(arow + j0 + kA + 4));
        v4i q2 = __builtin_nontemporal_load((const v4i*)(arow + j0 + 16 + kA));
        v4i q3 = __builtin_nontemporal_load((const v4i*)(arow + j0 + 16 + kA + 4));
        v4f g0 = *(const v4f*)(f2 + jbeg + j0 + kA);
        v4f g1 = *(const v4f*)(f2 + jbeg + j0 + kA + 4);
        v4f g2 = *(const v4f*)(f2 + jbeg + j0 + 16 + kA);
        v4f g3 = *(const v4f*)(f2 + jbeg + j0 + 16 + kA + 4);

        int   av[16];
        float fv[16];
#pragma unroll
        for (int t = 0; t < 4; ++t) {
            av[t] = q0[t]; av[4 + t] = q1[t]; av[8 + t] = q2[t]; av[12 + t] = q3[t];
            fv[t] = g0[t]; fv[4 + t] = g1[t]; fv[8 + t] = g2[t]; fv[12 + t] = g3[t];
        }

        v16h a;
#pragma unroll
        for (int t = 0; t < 16; ++t) {
            float p = gat_p(av[t], fv[t], f1r, mi);
            den += p;
            a[t] = (_Float16)p;
        }

        v8h b0 = *(const v8h*)(bcol + j0);
        v8h b1 = *(const v8h*)(bcol + j0 + 16);
        v16h b;
#pragma unroll
        for (int t = 0; t < 8; ++t) { b[t] = b0[t]; b[t + 8] = b1[t]; }

        c = __builtin_amdgcn_wmma_f32_16x16x32_f16(
                /*neg_a=*/false, a, /*neg_b=*/false, b,
                /*c_mod=*/(short)0, c, /*reuse_a=*/false, /*reuse_b=*/false);
    }

    // Partial row denominators: lane m and m+16 of wave 0 cover row m.
    __shared__ float sden[32];
    if (wave == 0) sden[lane] = den;
    __syncthreads();

    const size_t part = (size_t)blockIdx.x * GAT_SPLIT + blockIdx.y;
    float* atile = accbuf + part * (16 * GAT_OUTF);
    const int outc = c0 + laneL;
    // C layout: VGPR r -> (M = r + 8*hi, Ncol = laneL)
#pragma unroll
    for (int r = 0; r < 8; ++r) {
        int m = r + hi * 8;
        atile[m * GAT_OUTF + outc] = c[r];
    }
    if (wave == 0 && lane < 16)
        denbuf[part * 16 + lane] = sden[lane] + sden[lane + 16];
}

// ---------------------------------------------------------------------------
// Kernel 5: reduce K-split partials, divide, elu.
// ---------------------------------------------------------------------------
__global__ void __launch_bounds__(256)
gat_final_kernel(const float* __restrict__ accbuf, const float* __restrict__ denbuf,
                 float* __restrict__ out) {
    int idx = blockIdx.x * 256 + threadIdx.x;    // N*OUTF threads
    int row  = idx >> 6;
    int col  = idx & 63;
    int tile = row >> 4;
    int m    = row & 15;
    float acc = 0.f, den = 0.f;
#pragma unroll
    for (int s = 0; s < GAT_SPLIT; ++s) {
        size_t part = (size_t)tile * GAT_SPLIT + s;
        acc += accbuf[part * (16 * GAT_OUTF) + m * GAT_OUTF + col];
        den += denbuf[part * 16 + m];
    }
    den = fmaxf(den, 1e-20f);
    float v = acc / den;
    out[(size_t)row * GAT_OUTF + col] = (v > 0.f) ? v : (__expf(v) - 1.f);
}

// ---------------------------------------------------------------------------
extern "C" void kernel_launch(void* const* d_in, const int* in_sizes, int n_in,
                              void* d_out, int out_size, void* d_ws, size_t ws_size,
                              hipStream_t stream) {
    const float* inp = (const float*)d_in[0];   // [8192,128]
    const int*   adj = (const int*)  d_in[1];   // [8192,8192]
    const float* W   = (const float*)d_in[2];   // [128,64]
    const float* a   = (const float*)d_in[3];   // [128,1]
    float*       out = (float*)d_out;           // [8192,64]

    char* ws = (char*)d_ws;
    float*    h      = (float*)ws;                               // 2 MB
    _Float16* ht     = (_Float16*)(ws + 2u * 1024 * 1024);       // 1 MB
    float*    f1     = (float*)(ws + 3u * 1024 * 1024);          // 32 KB
    float*    f2     = (float*)(ws + 3u * 1024 * 1024 + 32768);  // 32 KB
    float*    gmax   = (float*)(ws + 3u * 1024 * 1024 + 65536);  // 4 B
    float*    accbuf = (float*)(ws + 4u * 1024 * 1024);          // 512*4*4 KB = 8 MB
    float*    denbuf = (float*)(ws + 12u * 1024 * 1024);         // 128 KB

    gat_h_kernel   <<<(GAT_N * GAT_OUTF) / 256, 256, 0, stream>>>(inp, W, h, ht);
    gat_f_kernel   <<<GAT_N / 256,              256, 0, stream>>>(h, a, f1, f2);
    gat_gmax_kernel<<<1,                        256, 0, stream>>>(f2, gmax);

    dim3 agrid(GAT_N / 16, GAT_SPLIT);
    gat_attn_kernel<<<agrid, 128, 0, stream>>>(adj, ht, f1, f2, gmax, accbuf, denbuf);

    gat_final_kernel<<<(GAT_N * GAT_OUTF) / 256, 256, 0, stream>>>(accbuf, denbuf, out);
}
